// EGNN_Full_27376121544708
// MI455X (gfx1250) — matrix operand order repaired and checked by reading, not compiled, not run on hardware
//
#include <hip/hip_runtime.h>

typedef _Float16 f16;
typedef __attribute__((ext_vector_type(16))) _Float16 v16h;
typedef __attribute__((ext_vector_type(8)))  _Float16 v8h;
typedef __attribute__((ext_vector_type(8)))  float    v8f;

#define DEPTH    5
#define HID      128
#define NODE_F   16
#define N_NODES  25000
#define N_EDGES  200000
#define N_GRAPHS 64
#define NT_MSG   (N_EDGES / 16)        // 12500 edge tiles
#define NT_UPD   ((N_NODES + 15) / 16) // 1563 node tiles

static __device__ __forceinline__ v8f wmma_f16(v16h a, v16h b, v8f c) {
  // v_wmma_f32_16x16x32_f16: D = A(16x32) x B(32x16) + C(16x16 f32)
  return __builtin_amdgcn_wmma_f32_16x16x32_f16(false, a, false, b, (short)0, c,
                                                false, false);
}

// Native global_atomic_add_f32 (no CAS-loop expansion).
static __device__ __forceinline__ void atomAddF32(float* p, float v) {
  unsafeAtomicAdd(p, v);
}

// ---------------------------------------------------------------------------
// Weight prep: f32 [depth][Ktot][128] -> f16 fragment-swizzled layout so that
// a B-fragment (32x16 f16) for (kk,nt) is 32 lanes x 16 contiguous halves.
// B layout (16-bit, 32x16): lanes 0-15 hold K=0..7,16..23, lanes
// 16-31 hold K=8..15,24..31; column N = lane%16.
// ---------------------------------------------------------------------------
__global__ void prep_swz(const float* __restrict__ W, f16* __restrict__ out,
                         int Ktot, int inLayerStride, int depth) {
  int tid = blockIdx.x * blockDim.x + threadIdx.x;
  int per = Ktot * HID;
  if (tid >= depth * per) return;
  int l = tid / per, r = tid % per;
  int f = r >> 9, L = (r >> 4) & 31, j = r & 15;
  int kk = f >> 3, nt = f & 7;
  int K = 32 * kk + ((L >= 16) ? 8 : 0) + (j & 7) + ((j >= 8) ? 16 : 0);
  int n = nt * 16 + (L & 15);
  out[tid] = (f16)W[(size_t)l * inLayerStride + K * HID + n];
}

__global__ void zero_f32(float* __restrict__ p, int n) {
  int t = blockIdx.x * blockDim.x + threadIdx.x;
  if (t < n) p[t] = 0.f;
}

__global__ void embed_kernel(const float* __restrict__ x,
                             const float* __restrict__ W,
                             const float* __restrict__ b,
                             float* __restrict__ h, f16* __restrict__ h16) {
  int t = blockIdx.x * blockDim.x + threadIdx.x;
  if (t >= N_NODES * HID) return;
  int i = t >> 7, j = t & 127;
  float a = b[j];
#pragma unroll
  for (int k = 0; k < NODE_F; ++k) a += x[i * NODE_F + k] * W[k * HID + j];
  h[t] = a;
  h16[t] = (f16)a;
}

__global__ void dist_kernel(const int* __restrict__ ei,
                            const float* __restrict__ pos,
                            float* __restrict__ dists) {
  int e = blockIdx.x * blockDim.x + threadIdx.x;
  if (e >= N_EDGES) return;
  int s = ei[e], d = ei[N_EDGES + e];
  float dx = pos[d * 3 + 0] - pos[s * 3 + 0];
  float dy = pos[d * 3 + 1] - pos[s * 3 + 1];
  float dz = pos[d * 3 + 2] - pos[s * 3 + 2];
  dists[e] = sqrtf(dx * dx + dy * dy + dz * dz);
}

__global__ void build_ucat(const f16* __restrict__ h16,
                           const float* __restrict__ agg,
                           f16* __restrict__ ucat) {
  int t = blockIdx.x * blockDim.x + threadIdx.x;
  if (t >= N_NODES * 256) return;
  int i = t >> 8, c = t & 255;
  ucat[t] = (c < HID) ? h16[i * HID + c] : (f16)agg[i * HID + c - HID];
}

// ---------------------------------------------------------------------------
// Message MLP over edges: one wave = 16 edges x 128 outputs.
// concat[h[dst], h[src], dist] @ W1 -> LN -> ReLU -> @ W2 -> LN -> ReLU
// -> atomic scatter-add into agg[dst].
// ---------------------------------------------------------------------------
__global__ __launch_bounds__(256) void msg_kernel(
    const f16* __restrict__ h16, const float* __restrict__ dists,
    const int* __restrict__ esrc, const int* __restrict__ edst,
    const f16* __restrict__ w1swz, const f16* __restrict__ w2swz,
    const float* __restrict__ w1row, const float* __restrict__ b1,
    const float* __restrict__ g1, const float* __restrict__ be1,
    const float* __restrict__ b2, const float* __restrict__ g2,
    const float* __restrict__ be2, float* __restrict__ agg) {
  __shared__ f16 sW1[256 * HID];      // 64 KB, fragment-swizzled
  __shared__ f16 sW2[HID * HID];      // 32 KB
  __shared__ f16 sM1[8][16 * HID];    // per-wave GEMM1->GEMM2 relayout, 32 KB
  __shared__ float sB1[HID], sG1[HID], sBe1[HID];
  __shared__ float sB2[HID], sG2[HID], sBe2[HID], sWr[HID];

  {
    const uint4* s1 = (const uint4*)w1swz;
    uint4* d1 = (uint4*)sW1;
    for (int i = threadIdx.x; i < 256 * HID / 8; i += 256) d1[i] = s1[i];
    const uint4* s2 = (const uint4*)w2swz;
    uint4* d2 = (uint4*)sW2;
    for (int i = threadIdx.x; i < HID * HID / 8; i += 256) d2[i] = s2[i];
    if (threadIdx.x < HID) {
      int t = threadIdx.x;
      sB1[t] = b1[t]; sG1[t] = g1[t]; sBe1[t] = be1[t];
      sB2[t] = b2[t]; sG2[t] = g2[t]; sBe2[t] = be2[t];
      sWr[t] = w1row[t];
    }
  }
  __syncthreads();

  const int wave = threadIdx.x >> 5;
  const int lane = threadIdx.x & 31;
  const int hi = lane >> 4;   // lane half (A: K offset; C/D: row +8)
  const int lm = lane & 15;   // A: row within tile; B/C/D: column
  f16* myM1 = sM1[wave];

  for (int tile = blockIdx.x * 8 + wave; tile < NT_MSG; tile += gridDim.x * 8) {
    const int e = tile * 16 + lm;
    const int idst = edst[e];
    const int isrc = esrc[e];
    const f16* pD = h16 + (size_t)idst * HID;
    const f16* pS = h16 + (size_t)isrc * HID;

    v8f acc[8] = {};
#pragma unroll
    for (int kk = 0; kk < 8; ++kk) {  // K = 0..255 (dst half then src half)
      const f16* p = (kk < 4) ? (pD + 32 * kk) : (pS + 32 * kk - HID);
      union { v16h v; v8h h[2]; } A;
      A.h[0] = *(const v8h*)(p + hi * 8);        // K = hi*8 .. +7
      A.h[1] = *(const v8h*)(p + hi * 8 + 16);   // K = 16+hi*8 .. +7
#pragma unroll
      for (int nt = 0; nt < 8; ++nt) {
        v16h B = *(const v16h*)(sW1 + ((kk * 8 + nt) * 32 + lane) * 16);
        acc[nt] = wmma_f16(A.v, B, acc[nt]);
      }
    }

    // distance column (K=256) + bias
    v8f dv = *(const v8f*)(dists + tile * 16 + hi * 8);
#pragma unroll
    for (int nt = 0; nt < 8; ++nt) {
      const int n = nt * 16 + lm;
      const float wr = sWr[n], bb = sB1[n];
#pragma unroll
      for (int r = 0; r < 8; ++r) acc[nt][r] += dv[r] * wr + bb;
    }

    // LayerNorm + ReLU; row m = r + hi*8 spans 16 lanes x 8 nt values
#pragma unroll
    for (int r = 0; r < 8; ++r) {
      float s = 0.f, s2 = 0.f;
#pragma unroll
      for (int nt = 0; nt < 8; ++nt) { float v = acc[nt][r]; s += v; s2 += v * v; }
#pragma unroll
      for (int mk = 1; mk < 16; mk <<= 1) { s += __shfl_xor(s, mk); s2 += __shfl_xor(s2, mk); }
      const float mean = s * (1.f / HID);
      const float rstd = rsqrtf(s2 * (1.f / HID) - mean * mean + 1e-5f);
#pragma unroll
      for (int nt = 0; nt < 8; ++nt) {
        const int n = nt * 16 + lm;
        float v = (acc[nt][r] - mean) * rstd * sG1[n] + sBe1[n];
        acc[nt][r] = fmaxf(v, 0.f);
      }
    }

    // relayout C-tiles -> A-fragments through per-wave LDS
#pragma unroll
    for (int nt = 0; nt < 8; ++nt) {
      const int n = nt * 16 + lm;
#pragma unroll
      for (int r = 0; r < 8; ++r) myM1[(r + hi * 8) * HID + n] = (f16)acc[nt][r];
    }
    asm volatile("s_wait_dscnt 0" ::: "memory");

    v8f acc2[8] = {};
#pragma unroll
    for (int kk = 0; kk < 4; ++kk) {
      const f16* p = myM1 + lm * HID + 32 * kk;
      union { v16h v; v8h h[2]; } A;
      A.h[0] = *(const v8h*)(p + hi * 8);
      A.h[1] = *(const v8h*)(p + hi * 8 + 16);
#pragma unroll
      for (int nt = 0; nt < 8; ++nt) {
        v16h B = *(const v16h*)(sW2 + ((kk * 8 + nt) * 32 + lane) * 16);
        acc2[nt] = wmma_f16(A.v, B, acc2[nt]);
      }
    }
#pragma unroll
    for (int nt = 0; nt < 8; ++nt) {
      const float bb = sB2[nt * 16 + lm];
#pragma unroll
      for (int r = 0; r < 8; ++r) acc2[nt][r] += bb;
    }
#pragma unroll
    for (int r = 0; r < 8; ++r) {
      float s = 0.f, s2 = 0.f;
#pragma unroll
      for (int nt = 0; nt < 8; ++nt) { float v = acc2[nt][r]; s += v; s2 += v * v; }
#pragma unroll
      for (int mk = 1; mk < 16; mk <<= 1) { s += __shfl_xor(s, mk); s2 += __shfl_xor(s2, mk); }
      const float mean = s * (1.f / HID);
      const float rstd = rsqrtf(s2 * (1.f / HID) - mean * mean + 1e-5f);
#pragma unroll
      for (int nt = 0; nt < 8; ++nt) {
        const int n = nt * 16 + lm;
        float v = (acc2[nt][r] - mean) * rstd * sG2[n] + sBe2[n];
        acc2[nt][r] = fmaxf(v, 0.f);
      }
    }

    // segment_sum over dst: L2-resident native f32 atomics
#pragma unroll
    for (int r = 0; r < 8; ++r) {
      const int m = r + hi * 8;
      const int node = __shfl(idst, m);  // row m's dst lives in lane m
      float* base = agg + (size_t)node * HID + lm;
#pragma unroll
      for (int nt = 0; nt < 8; ++nt) atomAddF32(base + nt * 16, acc2[nt][r]);
    }
  }
}

// ---------------------------------------------------------------------------
// Node update MLP: one wave = 16 nodes x 128 outputs; h += MLP(concat[h,agg]).
// ---------------------------------------------------------------------------
__global__ __launch_bounds__(256) void upd_kernel(
    const f16* __restrict__ ucat, const f16* __restrict__ w1swz,
    const f16* __restrict__ w2swz, const float* __restrict__ b1,
    const float* __restrict__ g1, const float* __restrict__ be1,
    const float* __restrict__ b2, const float* __restrict__ g2,
    const float* __restrict__ be2, float* __restrict__ h,
    f16* __restrict__ h16) {
  __shared__ f16 sW1[256 * HID];
  __shared__ f16 sW2[HID * HID];
  __shared__ f16 sM1[8][16 * HID];
  __shared__ float sB1[HID], sG1[HID], sBe1[HID];
  __shared__ float sB2[HID], sG2[HID], sBe2[HID];

  {
    const uint4* s1 = (const uint4*)w1swz;
    uint4* d1 = (uint4*)sW1;
    for (int i = threadIdx.x; i < 256 * HID / 8; i += 256) d1[i] = s1[i];
    const uint4* s2 = (const uint4*)w2swz;
    uint4* d2 = (uint4*)sW2;
    for (int i = threadIdx.x; i < HID * HID / 8; i += 256) d2[i] = s2[i];
    if (threadIdx.x < HID) {
      int t = threadIdx.x;
      sB1[t] = b1[t]; sG1[t] = g1[t]; sBe1[t] = be1[t];
      sB2[t] = b2[t]; sG2[t] = g2[t]; sBe2[t] = be2[t];
    }
  }
  __syncthreads();

  const int wave = threadIdx.x >> 5;
  const int lane = threadIdx.x & 31;
  const int hi = lane >> 4;
  const int lm = lane & 15;
  f16* myM1 = sM1[wave];

  for (int tile = blockIdx.x * 8 + wave; tile < NT_UPD; tile += gridDim.x * 8) {
    int nodeA = tile * 16 + lm;
    if (nodeA >= N_NODES) nodeA = N_NODES - 1;  // clamp; stores are masked
    const f16* p0 = ucat + (size_t)nodeA * 256;

    v8f acc[8] = {};
#pragma unroll
    for (int kk = 0; kk < 8; ++kk) {
      const f16* p = p0 + 32 * kk;
      union { v16h v; v8h h[2]; } A;
      A.h[0] = *(const v8h*)(p + hi * 8);
      A.h[1] = *(const v8h*)(p + hi * 8 + 16);
#pragma unroll
      for (int nt = 0; nt < 8; ++nt) {
        v16h B = *(const v16h*)(sW1 + ((kk * 8 + nt) * 32 + lane) * 16);
        acc[nt] = wmma_f16(A.v, B, acc[nt]);
      }
    }
#pragma unroll
    for (int nt = 0; nt < 8; ++nt) {
      const float bb = sB1[nt * 16 + lm];
#pragma unroll
      for (int r = 0; r < 8; ++r) acc[nt][r] += bb;
    }
#pragma unroll
    for (int r = 0; r < 8; ++r) {
      float s = 0.f, s2 = 0.f;
#pragma unroll
      for (int nt = 0; nt < 8; ++nt) { float v = acc[nt][r]; s += v; s2 += v * v; }
#pragma unroll
      for (int mk = 1; mk < 16; mk <<= 1) { s += __shfl_xor(s, mk); s2 += __shfl_xor(s2, mk); }
      const float mean = s * (1.f / HID);
      const float rstd = rsqrtf(s2 * (1.f / HID) - mean * mean + 1e-5f);
#pragma unroll
      for (int nt = 0; nt < 8; ++nt) {
        const int n = nt * 16 + lm;
        float v = (acc[nt][r] - mean) * rstd * sG1[n] + sBe1[n];
        acc[nt][r] = fmaxf(v, 0.f);
      }
    }

#pragma unroll
    for (int nt = 0; nt < 8; ++nt) {
      const int n = nt * 16 + lm;
#pragma unroll
      for (int r = 0; r < 8; ++r) myM1[(r + hi * 8) * HID + n] = (f16)acc[nt][r];
    }
    asm volatile("s_wait_dscnt 0" ::: "memory");

    v8f acc2[8] = {};
#pragma unroll
    for (int kk = 0; kk < 4; ++kk) {
      const f16* p = myM1 + lm * HID + 32 * kk;
      union { v16h v; v8h h[2]; } A;
      A.h[0] = *(const v8h*)(p + hi * 8);
      A.h[1] = *(const v8h*)(p + hi * 8 + 16);
#pragma unroll
      for (int nt = 0; nt < 8; ++nt) {
        v16h B = *(const v16h*)(sW2 + ((kk * 8 + nt) * 32 + lane) * 16);
        acc2[nt] = wmma_f16(A.v, B, acc2[nt]);
      }
    }
#pragma unroll
    for (int nt = 0; nt < 8; ++nt) {
      const float bb = sB2[nt * 16 + lm];
#pragma unroll
      for (int r = 0; r < 8; ++r) acc2[nt][r] += bb;
    }
#pragma unroll
    for (int r = 0; r < 8; ++r) {
      float s = 0.f, s2 = 0.f;
#pragma unroll
      for (int nt = 0; nt < 8; ++nt) { float v = acc2[nt][r]; s += v; s2 += v * v; }
#pragma unroll
      for (int mk = 1; mk < 16; mk <<= 1) { s += __shfl_xor(s, mk); s2 += __shfl_xor(s2, mk); }
      const float mean = s * (1.f / HID);
      const float rstd = rsqrtf(s2 * (1.f / HID) - mean * mean + 1e-5f);
#pragma unroll
      for (int nt = 0; nt < 8; ++nt) {
        const int n = nt * 16 + lm;
        float v = (acc2[nt][r] - mean) * rstd * sG2[n] + sBe2[n];
        acc2[nt][r] = fmaxf(v, 0.f);
      }
    }

    // residual: h += u (unique writer per element)
#pragma unroll
    for (int r = 0; r < 8; ++r) {
      const int node = tile * 16 + r + hi * 8;
      if (node < N_NODES) {
        size_t base = (size_t)node * HID + lm;
#pragma unroll
        for (int nt = 0; nt < 8; ++nt) {
          float nh = h[base + nt * 16] + acc2[nt][r];
          h[base + nt * 16] = nh;
          h16[base + nt * 16] = (f16)nh;
        }
      }
    }
  }
}

__global__ void pool_kernel(const float* __restrict__ h,
                            const int* __restrict__ batch,
                            float* __restrict__ gpool) {
  int t = blockIdx.x * blockDim.x + threadIdx.x;
  if (t >= N_NODES * HID) return;
  int i = t >> 7, j = t & 127;
  atomAddF32(&gpool[batch[i] * HID + j], h[t]);
}

__global__ __launch_bounds__(128) void pred_kernel(
    const float* __restrict__ gpool, const float* __restrict__ W1,
    const float* __restrict__ bb1, const float* __restrict__ W2,
    const float* __restrict__ bb2, float* __restrict__ out) {
  __shared__ float sh[HID];
  __shared__ float red[HID];
  int row = blockIdx.x, j = threadIdx.x;
  sh[j] = gpool[row * HID + j];
  __syncthreads();
  float a = bb1[j];
#pragma unroll 8
  for (int k = 0; k < HID; ++k) a += sh[k] * W1[k * HID + j];
  a = fmaxf(a, 0.f);
  red[j] = a * W2[j];
  __syncthreads();
  for (int s = 64; s > 0; s >>= 1) {
    if (j < s) red[j] += red[j + s];
    __syncthreads();
  }
  if (j == 0) out[row] = red[0] + bb2[0];
}

extern "C" void kernel_launch(void* const* d_in, const int* in_sizes, int n_in,
                              void* d_out, int out_size, void* d_ws,
                              size_t ws_size, hipStream_t stream) {
  (void)in_sizes; (void)n_in; (void)out_size; (void)ws_size;
  const float* x      = (const float*)d_in[0];
  const float* pos    = (const float*)d_in[1];
  const int*   ei     = (const int*)d_in[2];
  const int*   batch  = (const int*)d_in[3];
  const float* embW   = (const float*)d_in[4];
  const float* embB   = (const float*)d_in[5];
  const float* msgW1  = (const float*)d_in[6];
  const float* msgB1  = (const float*)d_in[7];
  const float* msgW2  = (const float*)d_in[8];
  const float* msgB2  = (const float*)d_in[9];
  const float* updW1  = (const float*)d_in[10];
  const float* updB1  = (const float*)d_in[11];
  const float* updW2  = (const float*)d_in[12];
  const float* updB2  = (const float*)d_in[13];
  const float* msgG1  = (const float*)d_in[14];
  const float* msgBe1 = (const float*)d_in[15];
  const float* msgG2  = (const float*)d_in[16];
  const float* msgBe2 = (const float*)d_in[17];
  const float* updG1  = (const float*)d_in[18];
  const float* updBe1 = (const float*)d_in[19];
  const float* updG2  = (const float*)d_in[20];
  const float* updBe2 = (const float*)d_in[21];
  const float* predW1 = (const float*)d_in[22];
  const float* predB1 = (const float*)d_in[23];
  const float* predW2 = (const float*)d_in[24];
  const float* predB2 = (const float*)d_in[25];

  char* w = (char*)d_ws;
  auto carve = [&](size_t bytes) {
    char* p = w;
    w += (bytes + 255) & ~(size_t)255;
    return p;
  };
  float* h     = (float*)carve((size_t)N_NODES * HID * 4);
  f16*   h16   = (f16*)carve((size_t)N_NODES * HID * 2);
  float* agg   = (float*)carve((size_t)N_NODES * HID * 4);
  f16*   ucat  = (f16*)carve((size_t)N_NODES * 256 * 2);
  float* dists = (float*)carve((size_t)N_EDGES * 4);
  f16* msgW1s  = (f16*)carve((size_t)DEPTH * 256 * HID * 2);
  f16* msgW2s  = (f16*)carve((size_t)DEPTH * HID * HID * 2);
  f16* updW1s  = (f16*)carve((size_t)DEPTH * 256 * HID * 2);
  f16* updW2s  = (f16*)carve((size_t)DEPTH * HID * HID * 2);
  float* gpool = (float*)carve((size_t)N_GRAPHS * HID * 4);

  const int* esrc = ei;
  const int* edst = ei + N_EDGES;

  prep_swz<<<(DEPTH * 256 * HID + 255) / 256, 256, 0, stream>>>(
      msgW1, msgW1s, 256, 257 * HID, DEPTH);
  prep_swz<<<(DEPTH * HID * HID + 255) / 256, 256, 0, stream>>>(
      msgW2, msgW2s, 128, HID * HID, DEPTH);
  prep_swz<<<(DEPTH * 256 * HID + 255) / 256, 256, 0, stream>>>(
      updW1, updW1s, 256, 256 * HID, DEPTH);
  prep_swz<<<(DEPTH * HID * HID + 255) / 256, 256, 0, stream>>>(
      updW2, updW2s, 128, HID * HID, DEPTH);

  embed_kernel<<<(N_NODES * HID + 255) / 256, 256, 0, stream>>>(x, embW, embB,
                                                                h, h16);
  dist_kernel<<<(N_EDGES + 255) / 256, 256, 0, stream>>>(ei, pos, dists);

  for (int l = 0; l < DEPTH; ++l) {
    zero_f32<<<(N_NODES * HID + 255) / 256, 256, 0, stream>>>(agg,
                                                              N_NODES * HID);
    msg_kernel<<<160, 256, 0, stream>>>(
        h16, dists, esrc, edst, msgW1s + (size_t)l * 256 * HID,
        msgW2s + (size_t)l * HID * HID,
        msgW1 + (size_t)l * 257 * HID + 256 * HID, msgB1 + l * HID,
        msgG1 + l * HID, msgBe1 + l * HID, msgB2 + l * HID, msgG2 + l * HID,
        msgBe2 + l * HID, agg);
    build_ucat<<<(N_NODES * 256 + 255) / 256, 256, 0, stream>>>(h16, agg, ucat);
    upd_kernel<<<120, 256, 0, stream>>>(
        ucat, updW1s + (size_t)l * 256 * HID, updW2s + (size_t)l * HID * HID,
        updB1 + l * HID, updG1 + l * HID, updBe1 + l * HID, updB2 + l * HID,
        updG2 + l * HID, updBe2 + l * HID, h, h16);
  }

  zero_f32<<<(N_GRAPHS * HID + 255) / 256, 256, 0, stream>>>(gpool,
                                                             N_GRAPHS * HID);
  pool_kernel<<<(N_NODES * HID + 255) / 256, 256, 0, stream>>>(h, batch, gpool);
  pred_kernel<<<N_GRAPHS, 128, 0, stream>>>(gpool, predW1, predB1, predW2,
                                            predB2, (float*)d_out);
}
